// Upnet_v3_MLMF_MSSR_CoarseToFine_61383672594888
// MI455X (gfx1250) — matrix operand order repaired and checked by reading, not compiled, not run on hardware
//
#include <hip/hip_runtime.h>
#include <math.h>

// Problem constants (from reference)
#define NPIX   1369      // 37*37
#define S_SRC  6
#define CINC   1536
#define COC    256
#define HIDN   384
#define DIC    512
#define DSN    16
#define XPWC   545       // 2*DS + DI + 1
#define OHN    148
#define OWN    148
#define OPIX   (OHN*OWN) // 21904
#define KCONV  2304      // 3*3*256
#define PW     150       // padded width/height for conv halo

typedef __attribute__((ext_vector_type(16))) _Float16 v16h;
typedef __attribute__((ext_vector_type(8)))  float    v8f;

__device__ __forceinline__ _Float16 f2h(float f){ return (_Float16)f; }

__device__ __forceinline__ void packf4(v16h& v, const int o, const float4 q){
  v[o+0]=f2h(q.x); v[o+1]=f2h(q.y); v[o+2]=f2h(q.z); v[o+3]=f2h(q.w);
}

// Raw fp32 A fragment (converted to f16 only at consumption time so the
// loads can be software-pipelined ahead of the conversion).
struct ARaw { float4 q0, q1, q2, q3; };

__device__ __forceinline__ ARaw loadA_raw(const float* p){
  const float4* ap = reinterpret_cast<const float4*>(p);
  ARaw r; r.q0 = ap[0]; r.q1 = ap[1]; r.q2 = ap[2]; r.q3 = ap[3];
  return r;
}
__device__ __forceinline__ v16h cvtA(const ARaw& r){
  v16h a;
  packf4(a, 0,  r.q0);
  packf4(a, 4,  r.q1);
  packf4(a, 8,  r.q2);
  packf4(a, 12, r.q3);
  return a;
}
__device__ __forceinline__ v16h loadB16(const _Float16* p){
  return *reinterpret_cast<const v16h*>(p);
}

#define WMMA_F16(A_, B_, C_) \
  __builtin_amdgcn_wmma_f32_16x16x32_f16(false, (A_), false, (B_), (short)0, (C_), false, false)

// ---------------------------------------------------------------------------
// WMMA GEMM:  C[M,N] = alpha * A[M,K] x BT^T  (+ C if accum)
//   A:  fp32 row-major (M,K), lda        (lda multiple of 4)
//   BT: f16  row-major (N,K), ldb        (pre-converted weights; ldb mult 16)
// K is a multiple of 64 here -> unroll-by-2 ping/pong pipeline with two
// disjoint fragment register sets (no rotation copies => partial waits).
// Load indices clamped (masked at store) -> branch-free inner loop.
// Block = 128 threads (4 waves); wave tile = 16x64; block tile = 64x64.
// ---------------------------------------------------------------------------
__global__ __launch_bounds__(128) void gemm_f16_wmma(
    const float* __restrict__ A, int lda,
    const _Float16* __restrict__ BT, int ldb,
    float* __restrict__ C, int ldc,
    int M, int N, int K,
    const float* __restrict__ alpha_ptr, int accum)
{
  const int lane = threadIdx.x & 31;
  const int wave = threadIdx.x >> 5;
  const int g    = lane >> 4;       // half-wave group
  const int ln   = lane & 15;
  const int row0 = blockIdx.y*64 + wave*16;
  const int col0 = blockIdx.x*64;

  const int r  = row0 + ln;
  const int rr = (r < M) ? r : (M-1);               // clamped A row
  const float* abase = A + (size_t)rr*lda + 16*g;

  const _Float16 *bb0, *bb1, *bb2, *bb3;
  {
    int n0 = col0 +  0 + ln; int c0 = (n0 < N) ? n0 : (N-1);
    int n1 = col0 + 16 + ln; int c1 = (n1 < N) ? n1 : (N-1);
    int n2 = col0 + 32 + ln; int c2 = (n2 < N) ? n2 : (N-1);
    int n3 = col0 + 48 + ln; int c3 = (n3 < N) ? n3 : (N-1);
    bb0 = BT + (size_t)c0*ldb + 16*g;
    bb1 = BT + (size_t)c1*ldb + 16*g;
    bb2 = BT + (size_t)c2*ldb + 16*g;
    bb3 = BT + (size_t)c3*ldb + 16*g;
  }

  v8f acc0 = {}, acc1 = {}, acc2 = {}, acc3 = {};

  // ping set (set0) @ kb=0
  ARaw ar0 = loadA_raw(abase);
  v16h b00 = loadB16(bb0), b01 = loadB16(bb1),
       b02 = loadB16(bb2), b03 = loadB16(bb3);

  int kb = 0;
  for (; kb + 64 < K; kb += 64) {
    // pong set (set1) @ kb+32 — issued before set0 is consumed
    ARaw ar1 = loadA_raw(abase + kb + 32);
    v16h b10 = loadB16(bb0 + kb + 32), b11 = loadB16(bb1 + kb + 32),
         b12 = loadB16(bb2 + kb + 32), b13 = loadB16(bb3 + kb + 32);
    {
      v16h a = cvtA(ar0);
      acc0 = WMMA_F16(a, b00, acc0);
      acc1 = WMMA_F16(a, b01, acc1);
      acc2 = WMMA_F16(a, b02, acc2);
      acc3 = WMMA_F16(a, b03, acc3);
    }
    // refill set0 @ kb+64 — issued before set1 is consumed
    ar0 = loadA_raw(abase + kb + 64);
    b00 = loadB16(bb0 + kb + 64); b01 = loadB16(bb1 + kb + 64);
    b02 = loadB16(bb2 + kb + 64); b03 = loadB16(bb3 + kb + 64);
    {
      v16h a = cvtA(ar1);
      acc0 = WMMA_F16(a, b10, acc0);
      acc1 = WMMA_F16(a, b11, acc1);
      acc2 = WMMA_F16(a, b12, acc2);
      acc3 = WMMA_F16(a, b13, acc3);
    }
  }
  // tail: exactly 64 columns of K remain (K % 64 == 0)
  {
    ARaw ar1 = loadA_raw(abase + kb + 32);
    v16h b10 = loadB16(bb0 + kb + 32), b11 = loadB16(bb1 + kb + 32),
         b12 = loadB16(bb2 + kb + 32), b13 = loadB16(bb3 + kb + 32);
    {
      v16h a = cvtA(ar0);
      acc0 = WMMA_F16(a, b00, acc0);
      acc1 = WMMA_F16(a, b01, acc1);
      acc2 = WMMA_F16(a, b02, acc2);
      acc3 = WMMA_F16(a, b03, acc3);
    }
    {
      v16h a = cvtA(ar1);
      acc0 = WMMA_F16(a, b10, acc0);
      acc1 = WMMA_F16(a, b11, acc1);
      acc2 = WMMA_F16(a, b12, acc2);
      acc3 = WMMA_F16(a, b13, acc3);
    }
  }

  const float alpha = alpha_ptr ? *alpha_ptr : 1.0f;
  v8f accs[4] = {acc0, acc1, acc2, acc3};
  #pragma unroll
  for (int t = 0; t < 4; ++t) {
    int col = col0 + t*16 + ln;
    if (col >= N) continue;
    #pragma unroll
    for (int j = 0; j < 8; ++j) {
      int row = row0 + j + 8*g;     // C/D layout: m = j + 8*(lane>>4)
      if (row < M) {
        float v = alpha * accs[t][j];
        if (accum) v += C[(size_t)row*ldc + col];
        C[(size_t)row*ldc + col] = v;
      }
    }
  }
}

// ---------------------------------------------------------------------------
// 3x3 SAME conv as implicit-im2col WMMA GEMM over a zero-haloed input.
//  upp: (150,150,256) fp32, zero border, interior = upsampled activations
//  WT:  (256,2304) f16  (co1_w transposed: co-major, k-contiguous)
//  out: (256, OPIX) fp32 (NCHW transposed output, +bias)
// K steps of 32 stay inside one (ky,kx) tap since 32 | 256. Same ping/pong
// unroll-by-2 pipeline as the GEMM (KCONV % 64 == 0).
// ---------------------------------------------------------------------------
__global__ __launch_bounds__(128) void conv3x3_wmma(
    const float* __restrict__ upp,
    const _Float16* __restrict__ WT,
    const float* __restrict__ bias,
    float* __restrict__ out)
{
  const int lane = threadIdx.x & 31;
  const int wave = threadIdx.x >> 5;
  const int g    = lane >> 4;
  const int ln   = lane & 15;
  const int row0 = blockIdx.y*64 + wave*16;
  const int col0 = blockIdx.x*64;

  const int p  = row0 + ln;
  const int pc = (p < OPIX) ? p : (OPIX-1);         // clamped pixel
  const int oy = pc / OWN, ox = pc % OWN;

  const _Float16* bb0 = WT + (size_t)(col0 +  0 + ln)*KCONV + 16*g;
  const _Float16* bb1 = WT + (size_t)(col0 + 16 + ln)*KCONV + 16*g;
  const _Float16* bb2 = WT + (size_t)(col0 + 32 + ln)*KCONV + 16*g;
  const _Float16* bb3 = WT + (size_t)(col0 + 48 + ln)*KCONV + 16*g;

  auto aaddr = [&](int kb) -> const float* {
    const int tap = kb >> 8;                        // 0..8, constant per frag
    const int iy  = oy + tap/3;                     // padded coords (0..149)
    const int ix  = ox + tap%3;
    return upp + ((size_t)iy*PW + ix)*COC + (kb & 255) + 16*g;
  };

  v8f acc0 = {}, acc1 = {}, acc2 = {}, acc3 = {};

  ARaw ar0 = loadA_raw(aaddr(0));
  v16h b00 = loadB16(bb0), b01 = loadB16(bb1),
       b02 = loadB16(bb2), b03 = loadB16(bb3);

  int kb = 0;
  for (; kb + 64 < KCONV; kb += 64) {
    ARaw ar1 = loadA_raw(aaddr(kb + 32));
    v16h b10 = loadB16(bb0 + kb + 32), b11 = loadB16(bb1 + kb + 32),
         b12 = loadB16(bb2 + kb + 32), b13 = loadB16(bb3 + kb + 32);
    {
      v16h a = cvtA(ar0);
      acc0 = WMMA_F16(a, b00, acc0);
      acc1 = WMMA_F16(a, b01, acc1);
      acc2 = WMMA_F16(a, b02, acc2);
      acc3 = WMMA_F16(a, b03, acc3);
    }
    ar0 = loadA_raw(aaddr(kb + 64));
    b00 = loadB16(bb0 + kb + 64); b01 = loadB16(bb1 + kb + 64);
    b02 = loadB16(bb2 + kb + 64); b03 = loadB16(bb3 + kb + 64);
    {
      v16h a = cvtA(ar1);
      acc0 = WMMA_F16(a, b10, acc0);
      acc1 = WMMA_F16(a, b11, acc1);
      acc2 = WMMA_F16(a, b12, acc2);
      acc3 = WMMA_F16(a, b13, acc3);
    }
  }
  {
    ARaw ar1 = loadA_raw(aaddr(kb + 32));
    v16h b10 = loadB16(bb0 + kb + 32), b11 = loadB16(bb1 + kb + 32),
         b12 = loadB16(bb2 + kb + 32), b13 = loadB16(bb3 + kb + 32);
    {
      v16h a = cvtA(ar0);
      acc0 = WMMA_F16(a, b00, acc0);
      acc1 = WMMA_F16(a, b01, acc1);
      acc2 = WMMA_F16(a, b02, acc2);
      acc3 = WMMA_F16(a, b03, acc3);
    }
    {
      v16h a = cvtA(ar1);
      acc0 = WMMA_F16(a, b10, acc0);
      acc1 = WMMA_F16(a, b11, acc1);
      acc2 = WMMA_F16(a, b12, acc2);
      acc3 = WMMA_F16(a, b13, acc3);
    }
  }

  v8f accs[4] = {acc0, acc1, acc2, acc3};
  #pragma unroll
  for (int t = 0; t < 4; ++t) {
    int co = col0 + t*16 + ln;
    float bv = bias[co];
    #pragma unroll
    for (int j = 0; j < 8; ++j) {
      int pr = row0 + j + 8*g;
      if (pr < OPIX) out[(size_t)co*OPIX + pr] = accs[t][j] + bv;
    }
  }
}

// ---------------------------------------------------------------------------
// Small helper kernels
// ---------------------------------------------------------------------------
__global__ void fill_zero(float* __restrict__ p, int n)
{
  int i = blockIdx.x*blockDim.x + threadIdx.x;
  if (i < n) p[i] = 0.f;
}

// (R,C) fp32 -> (C,R) f16
__global__ void transpose_h_kernel(const float* __restrict__ src,
                                   _Float16* __restrict__ dst, int R, int C)
{
  int i = blockIdx.x*blockDim.x + threadIdx.x;
  if (i >= R*C) return;
  int r = i / C, c = i % C;
  dst[(size_t)c*R + r] = (_Float16)src[i];
}

// straight fp32 -> f16 copy (layout preserved)
__global__ void cvt_h_kernel(const float* __restrict__ src,
                             _Float16* __restrict__ dst, int n)
{
  int i = blockIdx.x*blockDim.x + threadIdx.x;
  if (i < n) dst[i] = (_Float16)src[i];
}

// fm[s*CIN + c] = mean_n feats[s,n,c]
__global__ void featmean_kernel(const float* __restrict__ feats,
                                float* __restrict__ fm)
{
  int i = blockIdx.x*blockDim.x + threadIdx.x;
  if (i >= S_SRC*CINC) return;
  int s = i / CINC, c = i % CINC;
  const float* base = feats + (size_t)s*NPIX*CINC + c;
  float acc = 0.f;
  for (int n = 0; n < NPIX; ++n) acc += base[(size_t)n*CINC];
  fm[i] = acc * (1.f/NPIX);
}

// Tiny attention MLP + double softmax -> per-source weights wsrc[6].
__global__ void attn_kernel(const float* __restrict__ featmean,
                            const float* __restrict__ proj_w,
                            const float* __restrict__ w1,
                            const float* __restrict__ b1,
                            const float* __restrict__ w2,
                            const float* __restrict__ b2,
                            const float* __restrict__ src_scale,
                            float* __restrict__ wsrc)
{
  __shared__ float concat[S_SRC*COC];
  __shared__ float h[HIDN];
  __shared__ float logits[S_SRC];
  const int tid = threadIdx.x;

  for (int i = tid; i < S_SRC*COC; i += blockDim.x) {
    int s = i / COC, o = i % COC;
    const float* fm = &featmean[(size_t)s*CINC];
    const float* pw = &proj_w[((size_t)s*COC + o)*CINC];
    float acc = 0.f;
    for (int c = 0; c < CINC; ++c) acc += fm[c]*pw[c];
    concat[i] = acc;
  }
  __syncthreads();
  for (int j = tid; j < HIDN; j += blockDim.x) {
    float acc = b1[j];
    for (int i = 0; i < S_SRC*COC; ++i) acc += concat[i]*w1[(size_t)i*HIDN + j];
    h[j] = fmaxf(acc, 0.f);
  }
  __syncthreads();
  if (tid < S_SRC) {
    float acc = b2[tid];
    for (int j = 0; j < HIDN; ++j) acc += h[j]*w2[j*S_SRC + tid];
    logits[tid] = acc;
  }
  __syncthreads();
  if (tid == 0) {
    float w[S_SRC];
    float mx = -1e30f;
    for (int s = 0; s < S_SRC; ++s) mx = fmaxf(mx, logits[s]);
    float sum = 0.f;
    for (int s = 0; s < S_SRC; ++s) { w[s] = expf(logits[s]-mx); sum += w[s]; }
    for (int s = 0; s < S_SRC; ++s) w[s] = w[s]/sum * src_scale[s];
    mx = -1e30f;
    for (int s = 0; s < S_SRC; ++s) mx = fmaxf(mx, w[s]);
    sum = 0.f;
    for (int s = 0; s < S_SRC; ++s) { w[s] = expf(w[s]-mx); sum += w[s]; }
    for (int s = 0; s < S_SRC; ++s) wsrc[s] = w[s]/sum;
  }
}

// out[n,:] = in[rows-1-n,:]
__global__ void reverse_rows(const float* __restrict__ in,
                             float* __restrict__ out, int rows, int cols)
{
  int i = blockIdx.x*blockDim.x + threadIdx.x;
  if (i >= rows*cols) return;
  int n = i / cols, c = i % cols;
  out[i] = in[(size_t)(rows-1-n)*cols + c];
}

// Causal depthwise conv (K=4) + SiLU. z = zg[:, :512].
__global__ void dwconv_silu(const float* __restrict__ zg,
                            const float* __restrict__ cw,
                            const float* __restrict__ cb,
                            float* __restrict__ zc)
{
  int i = blockIdx.x*blockDim.x + threadIdx.x;
  if (i >= NPIX*DIC) return;
  int n = i / DIC, d = i % DIC;
  float acc = cb[d];
  #pragma unroll
  for (int k = 0; k < 4; ++k) {
    int m = n - 3 + k;
    if (m >= 0) acc += zg[(size_t)m*(2*DIC) + d] * cw[k*DIC + d];
  }
  zc[i] = acc / (1.f + expf(-acc));
}

// ---------------------------------------------------------------------------
// Selective scan: one lane per (channel d, state s); 16 lanes reduce over s
// each step via shfl_xor. yb[n,d] = (sum_s h*Cp + D*z) * silu(gate).
// grid = DI/16 blocks * 256 threads  (8 waves x 2 channels x 16 states)
// ---------------------------------------------------------------------------
__global__ __launch_bounds__(256) void scan_kernel(
    const float* __restrict__ zc,    // N x 512
    const float* __restrict__ xp,    // N x 545  [Bp(16) Cp(16) dlt(512) bias(1)]
    const float* __restrict__ zg,    // N x 1024 (gate at +512)
    const float* __restrict__ Alog,  // 512 x 16
    const float* __restrict__ Dp,    // 512
    float* __restrict__ yb)          // N x 512
{
  const int lane = threadIdx.x & 31;
  const int wave = threadIdx.x >> 5;
  const int s = lane & 15;
  const int d = blockIdx.x*16 + wave*2 + (lane >> 4);
  const float Acoef = -expf(Alog[d*DSN + s]);
  const float Dd = Dp[d];
  float h = 0.f;
  for (int n = 0; n < NPIX; ++n) {
    const float* xr = &xp[(size_t)n*XPWC];
    float draw  = xr[2*DSN + d] + xr[XPWC-1];
    float delta = (draw > 20.f) ? draw : log1pf(expf(draw));
    float z  = zc[(size_t)n*DIC + d];
    float dA = expf(delta * Acoef);
    h = dA*h + delta * xr[s] * z;
    float contrib = h * xr[DSN + s];
    contrib += __shfl_xor(contrib, 8, 16);
    contrib += __shfl_xor(contrib, 4, 16);
    contrib += __shfl_xor(contrib, 2, 16);
    contrib += __shfl_xor(contrib, 1, 16);
    if (s == 0) {
      float y  = contrib + Dd*z;
      float gt = zg[(size_t)n*(2*DIC) + DIC + d];
      yb[(size_t)n*DIC + d] = y * (gt / (1.f + expf(-gt)));
    }
  }
}

// Residual add + LayerNorm over CO=256 (one block per row).
__global__ void resln_kernel(const float* __restrict__ t,
                             const float* __restrict__ x,
                             const float* __restrict__ g,
                             const float* __restrict__ b,
                             float* __restrict__ out)
{
  __shared__ float sh[COC];
  const int n = blockIdx.x, c = threadIdx.x;
  float r = t[(size_t)n*COC + c] + x[(size_t)n*COC + c];
  sh[c] = r; __syncthreads();
  for (int s = COC/2; s > 0; s >>= 1) { if (c < s) sh[c] += sh[c+s]; __syncthreads(); }
  float mu = sh[0] * (1.f/COC);
  __syncthreads();
  float dv = r - mu; sh[c] = dv*dv; __syncthreads();
  for (int s = COC/2; s > 0; s >>= 1) { if (c < s) sh[c] += sh[c+s]; __syncthreads(); }
  float var = sh[0] * (1.f/COC);
  out[(size_t)n*COC + c] = g[c]*dv*rsqrtf(var + 1e-5f) + b[c];
}

// cat[n, 0:256] = fwd[n];  cat[n, 256:512] = bwd[N-1-n] (undo reversal)
__global__ void cat_kernel(const float* __restrict__ f,
                           const float* __restrict__ brev,
                           float* __restrict__ cat)
{
  int i = blockIdx.x*blockDim.x + threadIdx.x;
  if (i >= NPIX*2*COC) return;
  int n = i / (2*COC), c = i % (2*COC);
  cat[i] = (c < COC) ? f[(size_t)n*COC + c]
                     : brev[(size_t)(NPIX-1-n)*COC + (c - COC)];
}

// Half-pixel bilinear upsample 37x37 -> 148x148 (edge clamp) into the
// INTERIOR of the zero-haloed (150,150,256) buffer.
__global__ void upsample_kernel(const float* __restrict__ sp,
                                float* __restrict__ upp)
{
  int i = blockIdx.x*blockDim.x + threadIdx.x;
  if (i >= OPIX*COC) return;
  int c = i & (COC-1);
  int p = i >> 8;
  int ox = p % OWN, oy = p / OWN;
  float sy = (oy + 0.5f)*0.25f - 0.5f;
  float sx = (ox + 0.5f)*0.25f - 0.5f;
  int y0 = (int)floorf(sy); float fy = sy - (float)y0;
  int x0 = (int)floorf(sx); float fx = sx - (float)x0;
  int y0c = y0 < 0 ? 0 : (y0 > 36 ? 36 : y0);
  int y1t = y0 + 1; int y1c = y1t < 0 ? 0 : (y1t > 36 ? 36 : y1t);
  int x0c = x0 < 0 ? 0 : (x0 > 36 ? 36 : x0);
  int x1t = x0 + 1; int x1c = x1t < 0 ? 0 : (x1t > 36 ? 36 : x1t);
  float v00 = sp[((size_t)(y0c*37 + x0c))*COC + c];
  float v01 = sp[((size_t)(y0c*37 + x1c))*COC + c];
  float v10 = sp[((size_t)(y1c*37 + x0c))*COC + c];
  float v11 = sp[((size_t)(y1c*37 + x1c))*COC + c];
  upp[((size_t)(oy+1)*PW + (ox+1))*COC + c] =
      (1.f-fy)*((1.f-fx)*v00 + fx*v01) + fy*((1.f-fx)*v10 + fx*v11);
}

// ---------------------------------------------------------------------------
extern "C" void kernel_launch(void* const* d_in, const int* in_sizes, int n_in,
                              void* d_out, int out_size, void* d_ws, size_t ws_size,
                              hipStream_t stream)
{
  (void)in_sizes; (void)n_in; (void)out_size; (void)ws_size;

  const float* features = (const float*)d_in[0];
  const float* proj_w   = (const float*)d_in[1];
  const float* attn_w1  = (const float*)d_in[2];
  const float* attn_b1  = (const float*)d_in[3];
  const float* attn_w2  = (const float*)d_in[4];
  const float* attn_b2  = (const float*)d_in[5];
  const float* src_scale= (const float*)d_in[6];
  const float* in_w [2] = {(const float*)d_in[7],  (const float*)d_in[16]};
  const float* cnvw [2] = {(const float*)d_in[8],  (const float*)d_in[17]};
  const float* cnvb [2] = {(const float*)d_in[9],  (const float*)d_in[18]};
  const float* xp_w [2] = {(const float*)d_in[10], (const float*)d_in[19]};
  const float* Alog [2] = {(const float*)d_in[11], (const float*)d_in[20]};
  const float* Dp   [2] = {(const float*)d_in[12], (const float*)d_in[21]};
  const float* out_w[2] = {(const float*)d_in[13], (const float*)d_in[22]};
  const float* ln_g [2] = {(const float*)d_in[14], (const float*)d_in[23]};
  const float* ln_b [2] = {(const float*)d_in[15], (const float*)d_in[24]};
  const float* merge_w  = (const float*)d_in[25];
  const float* co1_w    = (const float*)d_in[26];
  const float* co1_b    = (const float*)d_in[27];

  // Workspace layout: fp32 region then f16 weight region (~57 MB total).
  // All fp32 offsets are multiples of 4 floats; f16 buffers are 16B aligned.
  float* ws     = (float*)d_ws;
  float* FM     = ws;                          // 6*1536
  float* WSRC   = FM + S_SRC*CINC;             // 16 (6 used)
  float* FUSED  = WSRC + 16;                   // N*256
  float* FUSEDR = FUSED  + (size_t)NPIX*COC;   // N*256
  float* ZG     = FUSEDR + (size_t)NPIX*COC;   // N*1024
  float* ZC     = ZG     + (size_t)NPIX*2*DIC; // N*512
  float* XPB    = ZC     + (size_t)NPIX*DIC;   // N*545 (+3 pad)
  float* YB     = XPB    + (size_t)NPIX*XPWC + 3; // N*512
  float* TT     = YB     + (size_t)NPIX*DIC;   // N*256
  float* OUTF   = TT     + (size_t)NPIX*COC;   // N*256
  float* OUTB   = OUTF   + (size_t)NPIX*COC;   // N*256
  float* CATB   = OUTB   + (size_t)NPIX*COC;   // N*512
  float* OUTM   = CATB   + (size_t)NPIX*2*COC; // N*256
  float* UPP    = OUTM   + (size_t)NPIX*COC;   // 150*150*256 (zero halo)
  // f16 weight region (each size is a multiple of 16 halves -> stays aligned)
  _Float16* PJH    = (_Float16*)(UPP + (size_t)PW*PW*COC);
  _Float16* WT_IN  = PJH    + (size_t)S_SRC*COC*CINC; // 1024 x 256
  _Float16* WT_XP  = WT_IN  + (size_t)2*DIC*COC;      // 545 x 512
  _Float16* WT_OUT = WT_XP  + (size_t)XPWC*DIC;       // 256 x 512
  _Float16* WT_MG  = WT_OUT + (size_t)COC*DIC;        // 256 x 512
  _Float16* WT_CV  = WT_MG  + (size_t)COC*DIC;        // 256 x 2304
  float* OUTD[2] = {OUTF, OUTB};

  dim3 blk(128);

  // 0) weight preprocessing: (K,N)->(N,K) + fp32->f16 (proj_w is already (N,K))
  cvt_h_kernel<<<(S_SRC*COC*CINC+255)/256, 256, 0, stream>>>(proj_w, PJH, S_SRC*COC*CINC);
  transpose_h_kernel<<<(KCONV*COC+255)/256, 256, 0, stream>>>(co1_w, WT_CV, KCONV, COC);
  transpose_h_kernel<<<(DIC*COC+255)/256, 256, 0, stream>>>(merge_w, WT_MG, DIC, COC);

  // 1) attention weights (tiny path)
  featmean_kernel<<<(S_SRC*CINC + 255)/256, 256, 0, stream>>>(features, FM);
  attn_kernel<<<1, 512, 0, stream>>>(FM, proj_w, attn_w1, attn_b1,
                                     attn_w2, attn_b2, src_scale, WSRC);

  // 2) fused[n,o] = sum_s w[s] * feats[s] @ proj_w[s]^T  (6 accumulating GEMMs)
  {
    dim3 grid((COC+63)/64, (NPIX+63)/64);
    for (int s = 0; s < S_SRC; ++s)
      gemm_f16_wmma<<<grid, blk, 0, stream>>>(
          features + (size_t)s*NPIX*CINC, CINC,
          PJH      + (size_t)s*COC*CINC,  CINC,
          FUSED, COC, NPIX, COC, CINC, WSRC + s, /*accum=*/(s > 0));
  }
  reverse_rows<<<(NPIX*COC + 255)/256, 256, 0, stream>>>(FUSED, FUSEDR, NPIX, COC);

  // 3) bidirectional SSM
  for (int dir = 0; dir < 2; ++dir) {
    const float* x = dir ? FUSEDR : FUSED;
    transpose_h_kernel<<<(COC*2*DIC+255)/256, 256, 0, stream>>>(in_w[dir], WT_IN, COC, 2*DIC);
    transpose_h_kernel<<<(DIC*XPWC+255)/256, 256, 0, stream>>>(xp_w[dir], WT_XP, DIC, XPWC);
    transpose_h_kernel<<<(DIC*COC+255)/256, 256, 0, stream>>>(out_w[dir], WT_OUT, DIC, COC);

    dim3 g1((2*DIC+63)/64, (NPIX+63)/64);
    gemm_f16_wmma<<<g1, blk, 0, stream>>>(x, COC, WT_IN, COC,
                                          ZG, 2*DIC, NPIX, 2*DIC, COC, nullptr, 0);
    dwconv_silu<<<(NPIX*DIC + 255)/256, 256, 0, stream>>>(ZG, cnvw[dir], cnvb[dir], ZC);
    dim3 g2((XPWC+63)/64, (NPIX+63)/64);
    gemm_f16_wmma<<<g2, blk, 0, stream>>>(ZC, DIC, WT_XP, DIC,
                                          XPB, XPWC, NPIX, XPWC, DIC, nullptr, 0);
    scan_kernel<<<DIC/16, 256, 0, stream>>>(ZC, XPB, ZG, Alog[dir], Dp[dir], YB);
    dim3 g3((COC+63)/64, (NPIX+63)/64);
    gemm_f16_wmma<<<g3, blk, 0, stream>>>(YB, DIC, WT_OUT, DIC,
                                          TT, COC, NPIX, COC, DIC, nullptr, 0);
    resln_kernel<<<NPIX, COC, 0, stream>>>(TT, x, ln_g[dir], ln_b[dir], OUTD[dir]);
  }

  // 4) concat (undoing bwd reversal) + merge GEMM
  cat_kernel<<<(NPIX*2*COC + 255)/256, 256, 0, stream>>>(OUTF, OUTB, CATB);
  {
    dim3 grid((COC+63)/64, (NPIX+63)/64);
    gemm_f16_wmma<<<grid, blk, 0, stream>>>(CATB, 2*COC, WT_MG, 2*COC,
                                            OUTM, COC, NPIX, COC, 2*COC, nullptr, 0);
  }

  // 5) zero halo, bilinear upsample, 3x3 conv (WMMA, transposed NCHW write)
  fill_zero<<<(PW*PW*COC + 255)/256, 256, 0, stream>>>(UPP, PW*PW*COC);
  upsample_kernel<<<(OPIX*COC + 255)/256, 256, 0, stream>>>(OUTM, UPP);
  {
    dim3 grid(COC/64, (OPIX+63)/64);
    conv3x3_wmma<<<grid, blk, 0, stream>>>(UPP, WT_CV, co1_b, (float*)d_out);
  }
}
